// LTE_49641232007338
// MI455X (gfx1250) — matrix operand order repaired and checked by reading, not compiled, not run on hardware
//
#include <hip/hip_runtime.h>
#include <hip/hip_bf16.h>

// ---------------------------------------------------------------------------
// LTE (Local Texture Estimator) forward for MI455X / gfx1250.
// Heavy compute (EDSR convs, coef/freq convs, 256-wide MLP) runs on
// v_wmma_f32_16x16x32_f16 with f16 storage + f32 accumulation.
//  - 64-ch feature maps use a zero-padded 66x66 pixel-major layout so conv
//    taps never diverge (no saveexec in the WMMA loop).
//  - WMMA A-operands (weights) are staged in LDS (ds_load_b128 feed).
// ---------------------------------------------------------------------------

typedef __attribute__((ext_vector_type(16))) _Float16 v16h;
typedef __attribute__((ext_vector_type(8)))  float    v8f;
typedef __attribute__((ext_vector_type(4)))  float    f4;

union Frag16 { v16h v; f4 q[2]; };
union HStore { _Float16 h[8]; f4 q; };

#define NPIX   4096          // 64*64
#define PW     66            // padded width (1-px halo)
#define PPIX   (PW*PW)       // 4356 padded pixels
#define CHN    256
#define NQ     65536
#define CHUNK_Q 8192
#define CHUNK_S (4*CHUNK_Q)  // 32768 samples per chunk
#define PI_F 3.14159265358979323846f

__device__ __forceinline__ int pad_idx(int py, int px) {
  return (py + 1) * PW + (px + 1);
}

// ---------------------------------------------------------------------------
// Zero fill (for padded-buffer halos; interior is overwritten by convs)
// ---------------------------------------------------------------------------
__global__ void zfill_kernel(_Float16* __restrict__ p, int n) {
  int i = blockIdx.x * 256 + threadIdx.x;
  if (i < n) p[i] = (_Float16)0.f;
}

// ---------------------------------------------------------------------------
// Weight repack: fp32 conv weights [OC][IC][3][3] -> f16 [tap][OC][IC]
// ---------------------------------------------------------------------------
__global__ void wprep_kernel(const float* __restrict__ src,
                             _Float16* __restrict__ dst, int OC, int IC) {
  int idx = blockIdx.x * 256 + threadIdx.x;
  int n = OC * IC * 9;
  if (idx >= n) return;
  int ic = idx % IC;
  int oc = (idx / IC) % OC;
  int t  = idx / (IC * OC);
  dst[idx] = (_Float16)src[(oc * IC + ic) * 9 + t];
}

// MLP weight repack: fp32 [K][O] -> f16 [O][K]  (o-major rows, contiguous K)
__global__ void mprep_kernel(const float* __restrict__ src,
                             _Float16* __restrict__ dst, int K, int O) {
  int idx = blockIdx.x * 256 + threadIdx.x;
  if (idx >= K * O) return;
  int k = idx % K;
  int o = idx / K;
  dst[o * K + k] = (_Float16)src[k * O + o];
}

// ---------------------------------------------------------------------------
// Input conv 3->64 (tiny): plain VALU, writes padded o1 and h (f16 [PPIX][64])
// ---------------------------------------------------------------------------
__global__ __launch_bounds__(256) void conv_in_kernel(
    const float* __restrict__ x, const float* __restrict__ w,
    const float* __restrict__ bias,
    _Float16* __restrict__ o1, _Float16* __restrict__ h) {
  __shared__ float ws[64 * 27];
  for (int i = threadIdx.x; i < 64 * 27; i += 256) ws[i] = w[i];
  __syncthreads();
  int p = blockIdx.x * 256 + threadIdx.x;
  int py = p >> 6, px = p & 63;
  float in[3][9];
#pragma unroll
  for (int c = 0; c < 3; ++c)
#pragma unroll
    for (int t = 0; t < 9; ++t) {
      int iy = py + t / 3 - 1, ix = px + t % 3 - 1;
      in[c][t] = ((unsigned)iy < 64u && (unsigned)ix < 64u)
                     ? x[c * NPIX + iy * 64 + ix] : 0.f;
    }
  const size_t pd = (size_t)pad_idx(py, px) * 64;
  for (int oc = 0; oc < 64; ++oc) {
    float a = bias[oc];
#pragma unroll
    for (int c = 0; c < 3; ++c)
#pragma unroll
      for (int t = 0; t < 9; ++t) a += ws[(oc * 3 + c) * 9 + t] * in[c][t];
    _Float16 hv = (_Float16)a;
    o1[pd + oc] = hv;
    h[pd + oc]  = hv;
  }
}

// ---------------------------------------------------------------------------
// WMMA conv3x3, 64 input channels, OC in {64,256}.
// fin/base are padded f16 [PPIX][64]; out is padded (64ch feats) or flat
// [NPIX][OC] (coef/freq). One wave = 16 oc x 16 pixel tile; 18 WMMAs.
// Weights for the block's (uniform) oc-tile staged in LDS: [9][16][64] f16.
// out = base + scale*(acc + bias);  optional ReLU.
// ---------------------------------------------------------------------------
__global__ __launch_bounds__(128) void conv3x3_wmma_kernel(
    const _Float16* __restrict__ fin, const _Float16* __restrict__ wt,
    const float* __restrict__ bias, const _Float16* __restrict__ base,
    float scale, int do_relu, _Float16* __restrict__ out, int OC,
    int out_padded) {
  extern __shared__ _Float16 wlds[];   // 9*16*64 halves = 18 KB
  const int lane = threadIdx.x & 31;
  const int wid  = threadIdx.x >> 5;
  const int tile = blockIdx.x * 4 + wid;
  const int pt   = tile & 255;          // NPIX/16 = 256 pixel tiles
  const int ot   = tile >> 8;           // uniform across the block (4 | 256)
  const int ln   = lane & 15;           // A-row (oc) and B-col (pixel)
  const int hi   = lane >> 4;           // half-wave id
  const int p    = pt * 16 + ln;
  const int py   = p >> 6, px = p & 63;

  // stage this oc-tile's weights: slice [9][16][64] from [9][OC][64]
  {
    const int otb = (blockIdx.x * 4) >> 8;   // == ot
    f4* dst = (f4*)wlds;
    const int nf4 = 9 * 16 * 64 / 8;         // 1152
    for (int i = threadIdx.x; i < nf4; i += 128) {
      int t = i >> 7;                        // / (16*64/8)
      int r = i & 127;
      const f4* src = (const f4*)(wt + (size_t)(t * OC + otb * 16) * 64);
      dst[i] = src[r];
    }
  }
  __syncthreads();

  v8f acc = (v8f)0.f;
#pragma unroll
  for (int t = 0; t < 9; ++t) {
    const int ky = t / 3 - 1, kx = t % 3 - 1;
    const _Float16* brow = fin + (size_t)pad_idx(py + ky, px + kx) * 64;
#pragma unroll
    for (int h2 = 0; h2 < 2; ++h2) {
      Frag16 a, b;
      // A from LDS: split-K layout
      const int icA = h2 * 32 + (hi ? 16 : 0);
      const _Float16* ap = wlds + ((t * 16 + ln) * 64) + icA;
      a.q[0] = *(const f4*)(ap);
      a.q[1] = *(const f4*)(ap + 8);
      // B from padded feature map: split-K layout, always in-bounds
      const int icB = h2 * 32 + (hi ? 16 : 0);
      const _Float16* bp = brow + icB;
      b.q[0] = *(const f4*)(bp);
      b.q[1] = *(const f4*)(bp + 8);
      acc = __builtin_amdgcn_wmma_f32_16x16x32_f16(
          false, a.v, false, b.v, (short)0, acc, false, false);
    }
  }
  // Epilogue: lane holds pixel p, oc = ot*16 + hi*8 + r (contiguous r)
  const int ocb = ot * 16 + (hi ? 8 : 0);
  const size_t pb = (size_t)pad_idx(py, px) * 64;
  HStore st;
#pragma unroll
  for (int r = 0; r < 8; ++r) {
    float v = scale * (acc[r] + bias[ocb + r]);
    if (base) v += (float)base[pb + ocb + r];
    if (do_relu) v = fmaxf(v, 0.f);
    st.h[r] = (_Float16)v;
  }
  if (out_padded) *(f4*)(out + pb + ocb) = st.q;
  else            *(f4*)(out + (size_t)p * OC + ocb) = st.q;
}

// ---------------------------------------------------------------------------
// Field construction: nearest gather of coef/freq, f = qf.rel + phase,
// X = qc * [cos(pi f), sin(pi f)] -> f16 [CHUNK_S][256].  Also areas.
// ---------------------------------------------------------------------------
__global__ __launch_bounds__(256) void field_kernel(
    const _Float16* __restrict__ coef, const _Float16* __restrict__ freq,
    const float* __restrict__ xcoord, const float* __restrict__ xcell,
    const float* __restrict__ phasew, int q0,
    _Float16* __restrict__ X, float* __restrict__ areas) {
  __shared__ float pw[256];
  for (int i = threadIdx.x; i < 256; i += 256) pw[i] = phasew[i];
  __syncthreads();
  int sl = blockIdx.x * 256 + threadIdx.x;   // 0..CHUNK_S-1
  int corner = sl >> 13;                     // / CHUNK_Q
  int qi = sl & (CHUNK_Q - 1);
  int q = q0 + qi;
  float vx = (corner < 2) ? -1.f : 1.f;      // order: (-1,-1)(-1,1)(1,-1)(1,1)
  float vy = (corner & 1) ? 1.f : -1.f;
  const float rr = 1.f / 64.f, eps = 1e-6f;
  float cy0 = xcoord[q * 2 + 0], cx0 = xcoord[q * 2 + 1];
  float cy = cy0 + vx * rr + eps, cx = cx0 + vy * rr + eps;
  cy = fminf(fmaxf(cy, -1.f + 1e-6f), 1.f - 1e-6f);
  cx = fminf(fmaxf(cx, -1.f + 1e-6f), 1.f - 1e-6f);
  int iy = (int)roundf(((cy + 1.f) * 64.f - 1.f) * 0.5f);
  int ix = (int)roundf(((cx + 1.f) * 64.f - 1.f) * 0.5f);
  iy = iy < 0 ? 0 : (iy > 63 ? 63 : iy);
  ix = ix < 0 ? 0 : (ix > 63 ? 63 : ix);
  int p = iy * 64 + ix;
  float rel0 = (cy0 - (-1.f + (2.f * iy + 1.f) / 64.f)) * 64.f;
  float rel1 = (cx0 - (-1.f + (2.f * ix + 1.f) / 64.f)) * 64.f;
  areas[corner * NQ + q] = fabsf(rel0 * rel1) + 1e-9f;
  float rc0 = xcell[q * 2 + 0] * 64.f, rc1 = xcell[q * 2 + 1] * 64.f;
  const _Float16* fr = freq + (size_t)p * CHN;
  const _Float16* cf = coef + (size_t)p * CHN;
  __builtin_prefetch(fr, 0, 3);   // global_prefetch_b8
  __builtin_prefetch(cf, 0, 3);
  _Float16* xo = X + (size_t)sl * CHN;
#pragma unroll 4
  for (int k = 0; k < 128; ++k) {
    float f = (float)fr[2 * k] * rel0 + (float)fr[2 * k + 1] * rel1 +
              rc0 * pw[2 * k] + rc1 * pw[2 * k + 1];
    float sv, cv;
    __sincosf(PI_F * f, &sv, &cv);
    xo[k]       = (_Float16)((float)cf[k] * cv);
    xo[128 + k] = (_Float16)((float)cf[128 + k] * sv);
  }
}

// ---------------------------------------------------------------------------
// MLP layer 256->256 (+bias, optional ReLU) with WMMA.
// One wave: 16 samples x 256 outputs (16 accumulator fragments).
// Full 128 KB f16 weight matrix staged in LDS per workgroup; A-fragments
// come from ds_load, B-fragments (activations) stream from HBM once.
// ---------------------------------------------------------------------------
__global__ __launch_bounds__(256) void mlp_layer_wmma_kernel(
    const _Float16* __restrict__ X, const _Float16* __restrict__ Wt,
    const float* __restrict__ bias, _Float16* __restrict__ Y, int do_relu) {
  extern __shared__ _Float16 wlds[];   // 256*256 halves = 128 KB
  const int lane  = threadIdx.x & 31;
  const int wid   = threadIdx.x >> 5;
  const int stile = blockIdx.x * 8 + wid;
  const int ln    = lane & 15;
  const int hi    = lane >> 4;
  const int s     = stile * 16 + ln;    // B column / D column: sample index

  __builtin_prefetch(X + (size_t)s * 256, 0, 1);  // warm activation row

  {
    f4* dst = (f4*)wlds;
    const f4* src = (const f4*)Wt;
    for (int i = threadIdx.x; i < 256 * 256 / 8; i += 256) dst[i] = src[i];
  }
  __syncthreads();

  v8f acc[16];
#pragma unroll
  for (int i = 0; i < 16; ++i) acc[i] = (v8f)0.f;

#pragma unroll
  for (int kk = 0; kk < 8; ++kk) {
    const int k0 = kk * 32;
    Frag16 b;  // activations: split-K, 16 contiguous halves per lane
    const _Float16* bp = X + (size_t)s * 256 + k0 + (hi ? 16 : 0);
    b.q[0] = *(const f4*)(bp);
    b.q[1] = *(const f4*)(bp + 8);
#pragma unroll
    for (int ot = 0; ot < 16; ++ot) {
      Frag16 a;  // weights from LDS: split-K
      const _Float16* ap = wlds + (ot * 16 + ln) * 256 + k0 + (hi ? 16 : 0);
      a.q[0] = *(const f4*)(ap);
      a.q[1] = *(const f4*)(ap + 8);
      acc[ot] = __builtin_amdgcn_wmma_f32_16x16x32_f16(
          false, a.v, false, b.v, (short)0, acc[ot], false, false);
    }
  }
#pragma unroll
  for (int ot = 0; ot < 16; ++ot) {
    const int ocb = ot * 16 + (hi ? 8 : 0);
    HStore st;
#pragma unroll
    for (int r = 0; r < 8; ++r) {
      float v = acc[ot][r] + bias[ocb + r];
      if (do_relu) v = fmaxf(v, 0.f);
      st.h[r] = (_Float16)v;
    }
    *(f4*)(Y + (size_t)s * 256 + ocb) = st.q;
  }
}

// ---------------------------------------------------------------------------
// Final MLP layer 256->3 (tiny): per-sample dots -> preds [4][Q][3]
// ---------------------------------------------------------------------------
__global__ __launch_bounds__(256) void final_layer_kernel(
    const _Float16* __restrict__ A, const float* __restrict__ w4,
    const float* __restrict__ b4, int q0, float* __restrict__ pred) {
  __shared__ float ws[768];
  __shared__ float bs[3];
  for (int i = threadIdx.x; i < 768; i += 256) ws[i] = w4[i];
  if (threadIdx.x < 3) bs[threadIdx.x] = b4[threadIdx.x];
  __syncthreads();
  int sl = blockIdx.x * 256 + threadIdx.x;
  int corner = sl >> 13, qi = sl & (CHUNK_Q - 1), q = q0 + qi;
  float a0 = bs[0], a1 = bs[1], a2 = bs[2];
  const _Float16* ar = A + (size_t)sl * 256;
#pragma unroll 4
  for (int k = 0; k < 256; ++k) {
    float v = (float)ar[k];
    a0 += v * ws[k * 3 + 0];
    a1 += v * ws[k * 3 + 1];
    a2 += v * ws[k * 3 + 2];
  }
  float* pr = pred + (size_t)(corner * NQ + q) * 3;
  pr[0] = a0; pr[1] = a1; pr[2] = a2;
}

// ---------------------------------------------------------------------------
// Combine: area-weighted corner blend (diagonal swap) + bilinear border of x
// ---------------------------------------------------------------------------
__global__ __launch_bounds__(256) void combine_kernel(
    const float* __restrict__ pred, const float* __restrict__ areas,
    const float* __restrict__ x, const float* __restrict__ xcoord,
    float* __restrict__ out) {
  int q = blockIdx.x * 256 + threadIdx.x;
  float a[4];
#pragma unroll
  for (int i = 0; i < 4; ++i) a[i] = areas[i * NQ + q];
  float tot = a[0] + a[1] + a[2] + a[3];
  float o0 = 0.f, o1 = 0.f, o2 = 0.f;
#pragma unroll
  for (int i = 0; i < 4; ++i) {
    float w = a[3 - i] / tot;   // diagonal swap
    const float* pr = pred + (size_t)(i * NQ + q) * 3;
    o0 += pr[0] * w; o1 += pr[1] * w; o2 += pr[2] * w;
  }
  float cy = xcoord[q * 2 + 0], cx = xcoord[q * 2 + 1];
  float fy = fminf(fmaxf(((cy + 1.f) * 64.f - 1.f) * 0.5f, 0.f), 63.f);
  float fx = fminf(fmaxf(((cx + 1.f) * 64.f - 1.f) * 0.5f, 0.f), 63.f);
  int y0 = (int)floorf(fy), x0 = (int)floorf(fx);
  float wy = fy - y0, wx = fx - x0;
  int y1 = y0 + 1 > 63 ? 63 : y0 + 1;
  int x1 = x0 + 1 > 63 ? 63 : x0 + 1;
#pragma unroll
  for (int c = 0; c < 3; ++c) {
    const float* xc = x + c * NPIX;
    float v = xc[y0 * 64 + x0] * (1 - wy) * (1 - wx) +
              xc[y0 * 64 + x1] * (1 - wy) * wx +
              xc[y1 * 64 + x0] * wy * (1 - wx) +
              xc[y1 * 64 + x1] * wy * wx;
    float o = (c == 0 ? o0 : (c == 1 ? o1 : o2)) + v;
    out[q * 3 + c] = o;
  }
}

// ---------------------------------------------------------------------------
// Host launcher.
// Input flattening (setup_inputs dict insertion order, depth-first):
//   0:x  1:x_coord  2:x_cell
//   3:c1_w 4:c1_b
//   5+4b .. : block b {w1,b1,w2,b2}  for b=0..15   (indices 5..68)
//   69:c2_w 70:c2_b 71:coef_w 72:coef_b 73:freq_w 74:freq_b 75:phase_w
//   76..83: mlp w0,b0,w1,b1,w2,b2,w3,b3
// ---------------------------------------------------------------------------
extern "C" void kernel_launch(void* const* d_in, const int* in_sizes, int n_in,
                              void* d_out, int out_size, void* d_ws, size_t ws_size,
                              hipStream_t stream) {
  const float* x       = (const float*)d_in[0];
  const float* xcoord  = (const float*)d_in[1];
  const float* xcell   = (const float*)d_in[2];
  const float* c1w     = (const float*)d_in[3];
  const float* c1b     = (const float*)d_in[4];
  const float* c2w     = (const float*)d_in[69];
  const float* c2b     = (const float*)d_in[70];
  const float* coefw   = (const float*)d_in[71];
  const float* coefb   = (const float*)d_in[72];
  const float* freqw   = (const float*)d_in[73];
  const float* freqb   = (const float*)d_in[74];
  const float* phasew  = (const float*)d_in[75];
  const float* mw[4] = {(const float*)d_in[76], (const float*)d_in[78],
                        (const float*)d_in[80], (const float*)d_in[82]};
  const float* mb[4] = {(const float*)d_in[77], (const float*)d_in[79],
                        (const float*)d_in[81], (const float*)d_in[83]};

  // workspace carve-out (256B aligned)
  size_t off = 0;
  auto carve = [&](size_t bytes) {
    void* p = (char*)d_ws + off;
    off += (bytes + 255) & ~(size_t)255;
    return p;
  };
  const size_t WN64  = (size_t)9 * 64 * 64;             // elems / 64->64 conv
  const size_t WSZ64 = WN64 * sizeof(_Float16);
  _Float16* wt_blk  = (_Float16*)carve(33 * WSZ64);     // 32 block convs + c2
  _Float16* wt_coef = (_Float16*)carve((size_t)9 * 256 * 64 * 2);
  _Float16* wt_freq = (_Float16*)carve((size_t)9 * 256 * 64 * 2);
  _Float16* wt_mlp0 = (_Float16*)carve((size_t)256 * 256 * 2);
  _Float16* wt_mlp1 = (_Float16*)carve((size_t)256 * 256 * 2);
  _Float16* wt_mlp2 = (_Float16*)carve((size_t)256 * 256 * 2);
  _Float16* o1buf   = (_Float16*)carve((size_t)PPIX * 64 * 2);  // padded
  _Float16* hbuf    = (_Float16*)carve((size_t)PPIX * 64 * 2);  // padded
  _Float16* tbuf    = (_Float16*)carve((size_t)PPIX * 64 * 2);  // padded
  _Float16* ffin    = (_Float16*)carve((size_t)PPIX * 64 * 2);  // padded
  _Float16* coefB   = (_Float16*)carve((size_t)NPIX * CHN * 2); // flat
  _Float16* freqB   = (_Float16*)carve((size_t)NPIX * CHN * 2); // flat
  _Float16* bufA    = (_Float16*)carve((size_t)CHUNK_S * CHN * 2);
  _Float16* bufB    = (_Float16*)carve((size_t)CHUNK_S * CHN * 2);
  float*    preds   = (float*)carve((size_t)4 * NQ * 3 * sizeof(float));
  float*    areas   = (float*)carve((size_t)4 * NQ * sizeof(float));
  (void)ws_size; (void)in_sizes; (void)n_in; (void)out_size;

  // ---- zero halo rings of padded feature buffers ----
  const int ZN = PPIX * 64;                   // 278784 = 1089*256
  zfill_kernel<<<(ZN + 255) / 256, 256, 0, stream>>>(o1buf, ZN);
  zfill_kernel<<<(ZN + 255) / 256, 256, 0, stream>>>(hbuf, ZN);
  zfill_kernel<<<(ZN + 255) / 256, 256, 0, stream>>>(tbuf, ZN);
  zfill_kernel<<<(ZN + 255) / 256, 256, 0, stream>>>(ffin, ZN);

  // ---- weight repack ----
  for (int b = 0; b < 16; ++b) {
    const float* w1 = (const float*)d_in[5 + 4 * b + 0];
    const float* w2 = (const float*)d_in[5 + 4 * b + 2];
    wprep_kernel<<<(WN64 + 255) / 256, 256, 0, stream>>>(w1, wt_blk + (2 * b) * WN64, 64, 64);
    wprep_kernel<<<(WN64 + 255) / 256, 256, 0, stream>>>(w2, wt_blk + (2 * b + 1) * WN64, 64, 64);
  }
  wprep_kernel<<<(WN64 + 255) / 256, 256, 0, stream>>>(c2w, wt_blk + 32 * WN64, 64, 64);
  const size_t WN256 = 9 * 256 * 64;
  wprep_kernel<<<(WN256 + 255) / 256, 256, 0, stream>>>(coefw, wt_coef, 256, 64);
  wprep_kernel<<<(WN256 + 255) / 256, 256, 0, stream>>>(freqw, wt_freq, 256, 64);
  mprep_kernel<<<(256 * 256 + 255) / 256, 256, 0, stream>>>(mw[0], wt_mlp0, 256, 256);
  mprep_kernel<<<(256 * 256 + 255) / 256, 256, 0, stream>>>(mw[1], wt_mlp1, 256, 256);
  mprep_kernel<<<(256 * 256 + 255) / 256, 256, 0, stream>>>(mw[2], wt_mlp2, 256, 256);

  // ---- EDSR encoder ----
  conv_in_kernel<<<NPIX / 256, 256, 0, stream>>>(x, c1w, c1b, o1buf, hbuf);
  const int CB64 = (NPIX / 16) * (64 / 16) / 4;    // 256 blocks of 4 waves
  const size_t CLDS = (size_t)9 * 16 * 64 * 2;     // 18 KB weight slice
  for (int b = 0; b < 16; ++b) {
    const float* b1 = (const float*)d_in[5 + 4 * b + 1];
    const float* b2 = (const float*)d_in[5 + 4 * b + 3];
    conv3x3_wmma_kernel<<<CB64, 128, CLDS, stream>>>(
        hbuf, wt_blk + (2 * b) * WN64, b1, nullptr, 1.f, 1, tbuf, 64, 1);
    conv3x3_wmma_kernel<<<CB64, 128, CLDS, stream>>>(
        tbuf, wt_blk + (2 * b + 1) * WN64, b2, hbuf, 0.1f, 0, hbuf, 64, 1);
  }
  conv3x3_wmma_kernel<<<CB64, 128, CLDS, stream>>>(
      hbuf, wt_blk + 32 * WN64, c2b, o1buf, 1.f, 0, ffin, 64, 1);

  const int CB256 = (NPIX / 16) * (256 / 16) / 4;  // 1024 blocks
  conv3x3_wmma_kernel<<<CB256, 128, CLDS, stream>>>(
      ffin, wt_coef, coefb, nullptr, 1.f, 0, coefB, 256, 0);
  conv3x3_wmma_kernel<<<CB256, 128, CLDS, stream>>>(
      ffin, wt_freq, freqb, nullptr, 1.f, 0, freqB, 256, 0);

  // ---- per-query field + MLP (chunked over queries) ----
  const int FBLK = CHUNK_S / 256;        // 128
  const int MBLK = (CHUNK_S / 16) / 8;   // 256 blocks of 8 waves
  const size_t MLDS = (size_t)256 * 256 * 2;  // 128 KB weight stage
  for (int chunk = 0; chunk < NQ / CHUNK_Q; ++chunk) {
    int q0 = chunk * CHUNK_Q;
    field_kernel<<<FBLK, 256, 0, stream>>>(coefB, freqB, xcoord, xcell,
                                           phasew, q0, bufA, areas);
    mlp_layer_wmma_kernel<<<MBLK, 256, MLDS, stream>>>(bufA, wt_mlp0, mb[0], bufB, 1);
    mlp_layer_wmma_kernel<<<MBLK, 256, MLDS, stream>>>(bufB, wt_mlp1, mb[1], bufA, 1);
    mlp_layer_wmma_kernel<<<MBLK, 256, MLDS, stream>>>(bufA, wt_mlp2, mb[2], bufB, 1);
    final_layer_kernel<<<FBLK, 256, 0, stream>>>(bufB, mw[3], mb[3], q0, preds);
  }

  // ---- combine + bilinear skip ----
  combine_kernel<<<NQ / 256, 256, 0, stream>>>(preds, areas, x, xcoord,
                                               (float*)d_out);
}
